// LocalSpectralTokenizer_70368744177845
// MI455X (gfx1250) — matrix operand order repaired and checked by reading, not compiled, not run on hardware
//
#include <hip/hip_runtime.h>
#include <hip/hip_bf16.h>
#include <math.h>
#include <stdint.h>

#define B_N 32
#define C_N 32
#define L_N 16384
#define NFR 511          // (16384-64)/32+1
#define NBIN 33
#define WIN 64
#define HOP 32
#define PEN 5.0f
#define ANCH 16
#define MAXB 544
#define MAXSEG 543
#define MAXT 2600
#define DEN_C 16383.0f
#define DEN_S 16384.0f
#define SEG_CHUNK 2048

typedef __attribute__((ext_vector_type(16))) _Float16 v16h;
typedef __attribute__((ext_vector_type(8)))  float    v8f;
typedef unsigned int v4u __attribute__((ext_vector_type(4)));
typedef int          v8i __attribute__((ext_vector_type(8)));
typedef int          v4i __attribute__((ext_vector_type(4)));

#if __has_builtin(__builtin_amdgcn_tensor_load_to_lds)
#define HAVE_TDM 1
#else
#define HAVE_TDM 0
#endif

// ---- CDNA5 async-to-LDS helpers (ASYNCcnt path) ----
__device__ __forceinline__ void async_ld_b32(unsigned int lds_off, const void* g) {
  asm volatile("global_load_async_to_lds_b32 %0, %1, off"
               :: "v"(lds_off), "v"((unsigned long long)(uintptr_t)g) : "memory");
}
__device__ __forceinline__ void async_ld_b128(unsigned int lds_off, const void* g) {
  asm volatile("global_load_async_to_lds_b128 %0, %1, off"
               :: "v"(lds_off), "v"((unsigned long long)(uintptr_t)g) : "memory");
}
__device__ __forceinline__ void wait_async0() {
  asm volatile("s_wait_asynccnt 0x0" ::: "memory");
}

// ---------------- Kernel A: channel mean (streaming, prefetch) ----------------
__global__ void k_mean(const float* __restrict__ x, float* __restrict__ sig) {
  int b = blockIdx.y;
  int l = blockIdx.x * 256 + threadIdx.x;
  const float* xb = x + (size_t)b * C_N * L_N + l;
  float acc = 0.f;
  for (int c = 0; c < C_N; ++c) {
    if (c + 1 < C_N) __builtin_prefetch(xb + (size_t)(c + 1) * L_N, 0, 0);
    acc += xb[(size_t)c * L_N];
  }
  sig[(size_t)b * L_N + l] = acc * (1.0f / C_N);
}

// ---------------- Kernel B: framed windowed DFT power -> log feats, via WMMA ----------------
// D(bins x frames) = W(bins x 64) * F(64 x frames), cos rows 0..32, sin rows 48..80.
// Signal strip staged into LDS by the Tensor Data Mover (1-D tile descriptor).
__global__ void __launch_bounds__(256) k_dft(const float* __restrict__ sig,
                                             float* __restrict__ feats) {
  __shared__ float    sch[4160];        // raw signal chunk (TDM destination)
  __shared__ _Float16 fr[128 * 64];     // windowed, mean-removed frames (f16)
  __shared__ _Float16 W[96 * 64];       // DFT matrix rows (cos 0..32, sin 48..80)
  __shared__ float    mns[128];
  const float TWO_PI = 6.283185307179586f;
  int b   = blockIdx.y;
  int f0  = blockIdx.x * 128;
  int tid = threadIdx.x;
  int nf_local = NFR - f0; if (nf_local > 128) nf_local = 128;
  int span = (nf_local - 1) * HOP + WIN;   // always <= L_N - f0*HOP
  int base = f0 * HOP;

#if HAVE_TDM
  if (tid < 32) {  // one wave issues the DMA; EXEC ignored, one TDM op per wave
    unsigned long long ga = (unsigned long long)(uintptr_t)(sig + (size_t)b * L_N + base);
    unsigned int ldsa = (unsigned int)(uintptr_t)(&sch[0]);
    v4u g0;
    g0[0] = 1u;                                        // count=1, user mode
    g0[1] = ldsa;                                      // lds_addr
    g0[2] = (unsigned int)(ga & 0xffffffffu);          // global_addr[31:0]
    g0[3] = (unsigned int)((ga >> 32) & 0x1ffffffu) | (2u << 30);  // addr[56:32] | type=2
    unsigned int sp = (unsigned int)span;
    v8i g1;
    g1[0] = (int)(2u << 16);                           // data_size=4B, no multicast
    g1[1] = (int)(sp << 16);                           // tensor_dim0[15:0]
    g1[2] = (int)((sp >> 16) | (1u << 16));            // tensor_dim0[31:16] | tensor_dim1=1
    g1[3] = (int)(sp << 16);                           // tile_dim0
    g1[4] = 1;                                         // tile_dim1=1, tile_dim2=0
    g1[5] = (int)sp;                                   // tensor_dim0_stride[31:0]
    g1[6] = (int)(sp << 16);                           // tensor_dim1_stride[15:0]
    g1[7] = (int)(sp >> 16);                           // tensor_dim1_stride[47:16]
    v4i z4 = {0, 0, 0, 0};
#if __clang_major__ >= 23
    v8i z8 = {0, 0, 0, 0, 0, 0, 0, 0};
    __builtin_amdgcn_tensor_load_to_lds(g0, g1, z4, z4, z8, 0);
#else
    __builtin_amdgcn_tensor_load_to_lds(g0, g1, z4, z4, 0);
#endif
    __builtin_amdgcn_s_wait_tensorcnt(0);
  }
#else
  for (int i = tid; i < span; i += 256) {
    int l = base + i;
    sch[i] = (l < L_N) ? sig[(size_t)b * L_N + l] : 0.f;
  }
#endif
  for (int i = tid; i < 96 * 64; i += 256) {
    int r = i >> 6, j = i & 63;
    float v = 0.f;
    if (r <= 32)                 v = __cosf(TWO_PI * (float)r        * (float)j / 64.0f);
    else if (r >= 48 && r <= 80) v = __sinf(TWO_PI * (float)(r - 48) * (float)j / 64.0f);
    W[i] = (_Float16)v;
  }
  __syncthreads();
  if (tid < 128) {
    float s = 0.f;
    int o = tid * HOP;
    for (int j = 0; j < WIN; ++j) s += sch[o + j];
    mns[tid] = s * (1.0f / WIN);
  }
  __syncthreads();
  for (int i = tid; i < 128 * 64; i += 256) {
    int f = i >> 6, j = i & 63;
    float h = 0.5f - 0.5f * __cosf(TWO_PI * (float)j / 63.0f);   // np.hanning(64)
    float v = (f < nf_local) ? (sch[f * HOP + j] - mns[f]) * h : 0.f;
    fr[i] = (_Float16)v;
  }
  __syncthreads();

  int wave = tid >> 5;
  int ln   = tid & 31;
  int half = ln >> 4;
  int lan  = ln & 15;
  int ntile = wave;                 // frames ntile*16 .. +15
  int nloc  = ntile * 16 + lan;

  v8f zero = {0.f,0.f,0.f,0.f,0.f,0.f,0.f,0.f};
  v8f acc[6];
  #pragma unroll
  for (int m = 0; m < 6; ++m) acc[m] = zero;

  #pragma unroll
  for (int m = 0; m < 6; ++m) {
    int rbase = (m < 3) ? m * 16 : 48 + (m - 3) * 16;
    int row = rbase + lan;          // A: lanes 0-15 and 16-31 both hold M = lane&15
    #pragma unroll
    for (int kc = 0; kc < 2; ++kc) {
      v16h a, bm;
      #pragma unroll
      for (int i = 0; i < 16; ++i) {
        int ka = ((i >> 3) << 4) + half * 8 + (i & 7);   // ISA 16-bit A 16x32 layout
        a[i]  = W[row * 64 + kc * 32 + ka];
        int kb = ((i >> 3) << 4) + half * 8 + (i & 7);   // B 32x16: same K order as A
        bm[i] = fr[nloc * 64 + kc * 32 + kb];
      }
      acc[m] = __builtin_amdgcn_wmma_f32_16x16x32_f16(false, a, false, bm,
                                                      (short)0, acc[m], false, false);
    }
  }
  // C layout: vgpr r -> (M = r + 8*half, N = lane&15)
  int t = f0 + ntile * 16 + lan;
  #pragma unroll
  for (int m = 0; m < 3; ++m) {
    #pragma unroll
    for (int r = 0; r < 8; ++r) {
      int bin = m * 16 + r + half * 8;
      if (bin < NBIN && t < NFR) {
        float cr = acc[m][r], si = acc[m + 3][r];
        feats[((size_t)b * NFR + t) * NBIN + bin] = __logf(cr * cr + si * si + 1e-8f);
      }
    }
  }
}

// ---------------- Kernel C: per-batch prefix sums over feats ----------------
__global__ void k_prefix(const float* __restrict__ feats,
                         float* __restrict__ cs, float* __restrict__ cs2) {
  int b = blockIdx.x;
  int d = threadIdx.x;
  if (d >= NBIN) return;
  size_t fo = (size_t)b * NFR * NBIN;
  size_t co = (size_t)b * (NFR + 1) * NBIN;
  float a1 = 0.f, a2 = 0.f;
  cs[co + d] = 0.f; cs2[co + d] = 0.f;
  for (int t = 1; t <= NFR; ++t) {
    float v = feats[fo + (size_t)(t - 1) * NBIN + d];
    a1 += v; a2 += v * v;
    cs[co + (size_t)t * NBIN + d]  = a1;
    cs2[co + (size_t)t * NBIN + d] = a2;
  }
}

// ---------------- Kernel D: PELT DP + backtrack -> boundaries ----------------
__global__ void __launch_bounds__(256) k_pelt(const float* __restrict__ cs,
                                              const float* __restrict__ cs2,
                                              int* __restrict__ bounds,
                                              int* __restrict__ nbounds) {
  __shared__ float Fc[NFR + 1];
  __shared__ int   prev[NFR + 1];
  __shared__ int   cand[NFR + 1];
  __shared__ float vals[NFR + 1];
  __shared__ int   rev[NFR + 1];
  __shared__ float rv[256];
  __shared__ int   ri[256];
  __shared__ int   ncand_s;
  int b = blockIdx.x;
  int tid = threadIdx.x;
  const float* CS  = cs  + (size_t)b * (NFR + 1) * NBIN;
  const float* CS2 = cs2 + (size_t)b * (NFR + 1) * NBIN;
  if (tid == 0) { Fc[0] = -PEN; prev[0] = 0; cand[0] = 0; ncand_s = 1; }
  __syncthreads();
  for (int t = 1; t <= NFR; ++t) {
    int nc = ncand_s;
    float bestv = INFINITY; int besti = 0x7fffffff;
    for (int ci = tid; ci < nc; ci += 256) {
      int c = cand[ci];
      float nn = (float)(t - c);
      float cost = 0.f;
      const float* t1 = CS  + (size_t)t * NBIN;
      const float* c1 = CS  + (size_t)c * NBIN;
      const float* t2 = CS2 + (size_t)t * NBIN;
      const float* c2 = CS2 + (size_t)c * NBIN;
      for (int d = 0; d < NBIN; ++d) {
        float s1 = t1[d] - c1[d];
        float s2 = t2[d] - c2[d];
        cost += s2 - s1 * s1 / nn;
      }
      float v = Fc[c] + cost + PEN;
      vals[ci] = v;
      if (v < bestv || (v == bestv && ci < besti)) { bestv = v; besti = ci; }
    }
    rv[tid] = bestv; ri[tid] = besti;
    __syncthreads();
    for (int s = 128; s > 0; s >>= 1) {
      if (tid < s) {
        float ov = rv[tid + s]; int oi = ri[tid + s];
        if (ov < rv[tid] || (ov == rv[tid] && oi < ri[tid])) { rv[tid] = ov; ri[tid] = oi; }
      }
      __syncthreads();
    }
    if (tid == 0) {
      Fc[t] = rv[0];
      prev[t] = cand[ri[0]];
      float thr = rv[0] + PEN;
      int nk = 0;
      for (int ci = 0; ci < nc; ++ci)
        if (vals[ci] <= thr) cand[nk++] = cand[ci];
      cand[nk++] = t;
      ncand_s = nk;
    }
    __syncthreads();
  }
  if (tid == 0) {
    int m = 0, t = NFR;
    while (t > 0 && m <= NFR) { rev[m++] = t; t = prev[t]; }
    int* bb = bounds + (size_t)b * MAXB;
    int nb = 0;
    bb[nb++] = 0;
    for (int i = m - 1; i >= 1; --i) {          // bps ascending, excluding last (=NFR)
      int k = rev[i];
      int bp = (k < NFR) ? k * HOP : L_N;
      if (bp > L_N) bp = L_N;
      if (bp - bb[nb - 1] >= 8 && nb < MAXB - 1) bb[nb++] = bp;
    }
    if (L_N - bb[nb - 1] < 8 && nb > 1) nb--;
    bb[nb++] = L_N;
    nbounds[b] = nb;
  }
}

// ---------------- Kernel E: per-segment spectrum stats -> patch length ----------------
// Segment samples staged chunk-wise into LDS with async-to-LDS loads (ASYNCcnt).
__global__ void __launch_bounds__(256) k_segstats(const float* __restrict__ sig,
                                                  const int* __restrict__ bounds,
                                                  const int* __restrict__ nbounds,
                                                  int* __restrict__ segpl,
                                                  float* __restrict__ segdp,
                                                  float* __restrict__ segbw) {
  __shared__ float spec[8192];
  __shared__ float chunk[SEG_CHUNK];
  __shared__ float red[256];
  __shared__ int   redi[256];
  int b  = blockIdx.y;
  int si = blockIdx.x;
  int nseg = nbounds[b] - 1;
  if (si >= nseg) return;
  int s = bounds[(size_t)b * MAXB + si];
  int e = bounds[(size_t)b * MAXB + si + 1];
  int n = e - s;
  int tid = threadIdx.x;
  float dp, bw;
  if (n < 4) {
    dp = (float)(n > 1 ? n : 1); bw = 0.f;
  } else {
    const float* xs = sig + (size_t)b * L_N + s;
    float a = 0.f;
    for (int j = tid; j < n; j += 256) a += xs[j];
    red[tid] = a; __syncthreads();
    for (int st = 128; st > 0; st >>= 1) { if (tid < st) red[tid] += red[tid + st]; __syncthreads(); }
    float mean = red[0] / (float)n;
    __syncthreads();
    int nf = n >> 1;
    const float TWO_PI = 6.283185307179586f;
    int nchunks = (n + SEG_CHUNK - 1) / SEG_CHUNK;
    int ngroups = (nf + 255) / 256;
    unsigned int ch_lds = (unsigned int)(uintptr_t)(&chunk[0]);
    for (int kg = 0; kg < ngroups; ++kg) {
      int k = 1 + kg * 256 + tid;          // this thread's bin (guarded at store)
      float ang = -TWO_PI * (float)k / (float)n;
      float cw = __cosf(ang), sw = __sinf(ang);
      float c = 1.f, ss = 0.f, re = 0.f, im = 0.f;
      for (int ch = 0; ch < nchunks; ++ch) {
        int off0 = ch * SEG_CHUNK;
        int clen = n - off0; if (clen > SEG_CHUNK) clen = SEG_CHUNK;
        __syncthreads();                   // previous chunk fully consumed
        int c4 = clen & ~3;
        for (int i = tid * 4; i < c4; i += 1024)
          async_ld_b128(ch_lds + (unsigned int)i * 4u, xs + off0 + i);
        for (int i = c4 + tid; i < clen; i += 256)
          async_ld_b32(ch_lds + (unsigned int)i * 4u, xs + off0 + i);
        wait_async0();
        __syncthreads();                   // chunk staged for all waves
        for (int j = 0; j < clen; ++j) {
          float xv = chunk[j] - mean;
          re += xv * c; im += xv * ss;
          float c2 = c * cw - ss * sw;
          ss = c * sw + ss * cw;
          c = c2;
        }
      }
      if (k <= nf) spec[k - 1] = re * re + im * im;
    }
    __syncthreads();
    float a1 = 0.f;
    for (int i = tid; i < nf; i += 256) a1 += spec[i];
    red[tid] = a1; __syncthreads();
    for (int st = 128; st > 0; st >>= 1) { if (tid < st) red[tid] += red[tid + st]; __syncthreads(); }
    float tot = red[0];
    __syncthreads();
    if (tot <= 0.f) {
      dp = (float)n; bw = 0.f;
    } else {
      float mv = -INFINITY; int mi = 0x7fffffff;
      for (int i = tid; i < nf; i += 256) {
        float v = spec[i];
        if (v > mv || (v == mv && i < mi)) { mv = v; mi = i; }
      }
      red[tid] = mv; redi[tid] = mi; __syncthreads();
      for (int st = 128; st > 0; st >>= 1) {
        if (tid < st) {
          float ov = red[tid + st]; int oi = redi[tid + st];
          if (ov > red[tid] || (ov == red[tid] && oi < redi[tid])) { red[tid] = ov; redi[tid] = oi; }
        }
        __syncthreads();
      }
      int kk = redi[0] + 1;
      dp = (float)n / (float)kk;
      __syncthreads();
      float inv_tot = 1.f / tot;
      float invnf = 1.f / (float)nf;
      float a2 = 0.f;
      for (int i = tid; i < nf; i += 256) a2 += spec[i] * inv_tot * ((float)(i + 1) * invnf);
      red[tid] = a2; __syncthreads();
      for (int st = 128; st > 0; st >>= 1) { if (tid < st) red[tid] += red[tid + st]; __syncthreads(); }
      float cc = red[0];
      __syncthreads();
      float a3 = 0.f;
      for (int i = tid; i < nf; i += 256) {
        float df = (float)(i + 1) * invnf - cc;
        a3 += spec[i] * inv_tot * df * df;
      }
      red[tid] = a3; __syncthreads();
      for (int st = 128; st > 0; st >>= 1) { if (tid < st) red[tid] += red[tid + st]; __syncthreads(); }
      bw = sqrtf(red[0]);
    }
  }
  if (tid == 0) {
    float raw = dp / (1.0f + bw);                 // GEN_ALPHA=GEN_BETA=1
    int pl = (int)(2.0f * rintf(raw * 0.5f));     // PATCH_GRID=2, banker's rounding
    if (pl < 8) pl = 8;
    if (pl > 64) pl = 64;
    segpl[(size_t)b * MAXSEG + si] = pl;
    segdp[(size_t)b * MAXSEG + si] = dp;
    segbw[(size_t)b * MAXSEG + si] = bw;
  }
}

// ---------------- Kernel F: token list build + max_tok ----------------
__global__ void k_tokens(const int* __restrict__ bounds, const int* __restrict__ nbounds,
                         const int* __restrict__ segpl, const float* __restrict__ segdp,
                         const float* __restrict__ segbw,
                         int* __restrict__ tst, int* __restrict__ ten, float* __restrict__ treg,
                         int* __restrict__ ntok, int* __restrict__ maxtok) {
  __shared__ int nt[B_N];
  int b = threadIdx.x;
  if (b < B_N) {
    const int* bb = bounds + (size_t)b * MAXB;
    int nseg = nbounds[b] - 1;
    int tok = 0;
    for (int i = 0; i < nseg; ++i) {
      int s = bb[i], e = bb[i + 1];
      if (e <= s) continue;
      int pl = segpl[(size_t)b * MAXSEG + i];
      float dp = segdp[(size_t)b * MAXSEG + i];
      float bw = segbw[(size_t)b * MAXSEG + i];
      int seg_len = e - s;
      float r0 = dp / DEN_S, r2 = (float)seg_len / DEN_S;
      int n_full = seg_len / pl;
      for (int j = 0; j < n_full && tok < MAXT; ++j) {
        int st = s + j * pl;
        tst[(size_t)b * MAXT + tok] = st;
        ten[(size_t)b * MAXT + tok] = st + pl;
        float* rg = treg + ((size_t)b * MAXT + tok) * 3;
        rg[0] = r0; rg[1] = bw; rg[2] = r2;
        tok++;
      }
      if (n_full * pl < seg_len && tok < MAXT) {
        tst[(size_t)b * MAXT + tok] = s + n_full * pl;
        ten[(size_t)b * MAXT + tok] = e;
        float* rg = treg + ((size_t)b * MAXT + tok) * 3;
        rg[0] = r0; rg[1] = bw; rg[2] = r2;
        tok++;
      }
    }
    ntok[b] = tok;
    nt[b] = tok;
  }
  __syncthreads();
  if (threadIdx.x == 0) {
    int m = 0;
    for (int i = 0; i < B_N; ++i) if (nt[i] > m) m = nt[i];
    maxtok[0] = m;
  }
}

// ---------------- Kernel G: zero output ----------------
__global__ void k_zero(float* __restrict__ out, int n) {
  int i = blockIdx.x * 256 + threadIdx.x;
  if (i < n) out[i] = 0.f;
}

// ---------------- Kernel H1: metadata outputs ----------------
__global__ void k_meta(const int* __restrict__ tst, const int* __restrict__ ten,
                       const float* __restrict__ treg, const int* __restrict__ ntok,
                       const int* __restrict__ maxtok, float* __restrict__ out,
                       long long out_n) {
  int MT = maxtok[0];
  if (MT <= 0) return;
  long long idx = (long long)blockIdx.x * 256 + threadIdx.x;
  long long total = (long long)B_N * MT;
  if (idx >= total) return;
  int b = (int)(idx / MT);
  int t = (int)(idx % MT);
  long long O1 = (long long)B_N * MT * C_N * ANCH;
  long long O2 = O1 + total;
  long long O3 = O2 + total;
  long long O4 = O3 + total;
  long long O5 = O4 + total;
  long long O6 = O5 + total;
  long long O7 = O6 + 3 * total;
  int n = ntok[b];
  if (t < n) {
    int st = tst[(size_t)b * MAXT + t];
    int en = ten[(size_t)b * MAXT + t];
    const float* rg = treg + ((size_t)b * MAXT + t) * 3;
    long long p;
    p = O1 + idx; if (p < out_n) out[p] = 1.0f;
    p = O2 + idx; if (p < out_n) out[p] = (float)st;
    p = O3 + idx; if (p < out_n) out[p] = (float)en;
    p = O4 + idx; if (p < out_n) out[p] = ((float)st + (float)en - 1.0f) * 0.5f / DEN_C;
    p = O5 + idx; if (p < out_n) out[p] = (float)(en - st) / DEN_S;
    p = O6 + idx * 3; if (p + 2 < out_n) { out[p] = rg[0]; out[p + 1] = rg[1]; out[p + 2] = rg[2]; }
  }
  if (t == 0) { long long p = O7 + b; if (p < out_n) out[p] = (float)n; }
}

// ---------------- Kernel H2: patch extraction with linear resize ----------------
__global__ void __launch_bounds__(256) k_patches(const float* __restrict__ x,
                                                 const int* __restrict__ tst,
                                                 const int* __restrict__ ten,
                                                 const int* __restrict__ ntok,
                                                 const int* __restrict__ maxtok,
                                                 float* __restrict__ out, long long out_n) {
  int b = blockIdx.y;
  int t = blockIdx.x;
  int MT = maxtok[0];
  if (t >= MT || t >= ntok[b]) return;
  int st = tst[(size_t)b * MAXT + t];
  int en = ten[(size_t)b * MAXT + t];
  int inlen = en - st;
  const float* xb = x + (size_t)b * C_N * L_N;
  float scale = (float)inlen / (float)ANCH;
  for (int idx = threadIdx.x; idx < C_N * ANCH; idx += 256) {
    int c = idx >> 4;
    int i = idx & 15;
    float val;
    if (inlen == ANCH) {
      val = xb[(size_t)c * L_N + st + i];
    } else {
      float coord = ((float)i + 0.5f) * scale - 0.5f;
      if (coord < 0.f) coord = 0.f;
      float mx = (float)(inlen - 1);
      if (coord > mx) coord = mx;
      int lo = (int)floorf(coord);
      int hi = lo + 1; if (hi > inlen - 1) hi = inlen - 1;
      float w = coord - (float)lo;
      float x0 = xb[(size_t)c * L_N + st + lo];
      float x1 = xb[(size_t)c * L_N + st + hi];
      val = x0 * (1.f - w) + x1 * w;
    }
    long long p = (((long long)b * MT + t) * C_N + c) * ANCH + i;
    if (p < out_n) out[p] = val;
  }
}

extern "C" void kernel_launch(void* const* d_in, const int* in_sizes, int n_in,
                              void* d_out, int out_size, void* d_ws, size_t ws_size,
                              hipStream_t stream) {
  (void)in_sizes; (void)n_in; (void)ws_size;
  const float* x = (const float*)d_in[0];
  float* out = (float*)d_out;
  char* ws = (char*)d_ws;
  size_t off = 0;
  auto take = [&](size_t bytes) { size_t r = off; off += (bytes + 255) & ~(size_t)255; return r; };
  float* sig   = (float*)(ws + take((size_t)B_N * L_N * 4));
  float* feats = (float*)(ws + take((size_t)B_N * NFR * NBIN * 4));
  float* cs    = (float*)(ws + take((size_t)B_N * (NFR + 1) * NBIN * 4));
  float* cs2   = (float*)(ws + take((size_t)B_N * (NFR + 1) * NBIN * 4));
  int*   bounds  = (int*)(ws + take((size_t)B_N * MAXB * 4));
  int*   nbounds = (int*)(ws + take((size_t)B_N * 4));
  int*   segpl   = (int*)(ws + take((size_t)B_N * MAXSEG * 4));
  float* segdp   = (float*)(ws + take((size_t)B_N * MAXSEG * 4));
  float* segbw   = (float*)(ws + take((size_t)B_N * MAXSEG * 4));
  int*   tst     = (int*)(ws + take((size_t)B_N * MAXT * 4));
  int*   ten     = (int*)(ws + take((size_t)B_N * MAXT * 4));
  float* treg    = (float*)(ws + take((size_t)B_N * MAXT * 12));
  int*   ntok    = (int*)(ws + take((size_t)B_N * 4));
  int*   maxtok  = (int*)(ws + take(4));

  k_mean<<<dim3(L_N / 256, B_N), 256, 0, stream>>>(x, sig);
  k_dft<<<dim3(4, B_N), 256, 0, stream>>>(sig, feats);
  k_prefix<<<dim3(B_N), 64, 0, stream>>>(feats, cs, cs2);
  k_pelt<<<dim3(B_N), 256, 0, stream>>>(cs, cs2, bounds, nbounds);
  k_segstats<<<dim3(MAXSEG, B_N), 256, 0, stream>>>(sig, bounds, nbounds, segpl, segdp, segbw);
  k_tokens<<<1, 32, 0, stream>>>(bounds, nbounds, segpl, segdp, segbw,
                                 tst, ten, treg, ntok, maxtok);
  int zb = (out_size + 255) / 256;
  if (zb > 0) k_zero<<<zb, 256, 0, stream>>>(out, out_size);
  long long mb = ((long long)B_N * MAXT + 255) / 256;
  k_meta<<<(int)mb, 256, 0, stream>>>(tst, ten, treg, ntok, maxtok, out, (long long)out_size);
  k_patches<<<dim3(MAXT, B_N), 256, 0, stream>>>(x, tst, ten, ntok, maxtok, out, (long long)out_size);
}
